// VectorQuantizer_43035572306092
// MI455X (gfx1250) — matrix operand order repaired and checked by reading, not compiled, not run on hardware
//
#include <hip/hip_runtime.h>
#include <hip/hip_bf16.h>

typedef __attribute__((ext_vector_type(8)))  __bf16 v8bf;
typedef __attribute__((ext_vector_type(16))) __bf16 v16bf;
typedef __attribute__((ext_vector_type(8)))  float  v8f;

// Problem constants (from reference setup_inputs)
constexpr int B  = 16;
constexpr int D  = 256;
constexpr int HW = 1024;               // H*W
constexpr int N  = B * HW;             // 16384 query vectors
constexpr int K  = 8192;               // codebook size
constexpr int OUT_ELEMS = B * D * HW;  // 4,194,304
constexpr int LOSS_POS = OUT_ELEMS;
constexpr int IDX_POS  = OUT_ELEMS + 1;

// LDS row stride in bf16 elements: 264*2B = 528B = 132 banks -> consecutive
// rows offset by 4 banks, so 16-row column reads (4 banks wide) are conflict
// free; 528 % 16 == 0 keeps every b128 chunk 16-byte aligned.
constexpr int LSB = 264;

__device__ __forceinline__ v16bf cat8(v8bf a, v8bf b) {
    return __builtin_shufflevector(a, b, 0,1,2,3,4,5,6,7,8,9,10,11,12,13,14,15);
}

// ---------------------------------------------------------------------------
// Kernel 1: ee[k] = ||emb_k||^2 ; block 0 zeroes the loss accumulator slot.
// ---------------------------------------------------------------------------
__global__ void vq_ee_kernel(const float* __restrict__ emb,
                             float* __restrict__ ee,
                             float* __restrict__ out) {
    const int k = blockIdx.x;
    const int t = threadIdx.x;          // 0..255 == dim
    float v = emb[k * D + t];
    float s = v * v;
    #pragma unroll
    for (int m = 16; m >= 1; m >>= 1) s += __shfl_xor(s, m, 32);
    __shared__ float red[8];
    if ((t & 31) == 0) red[t >> 5] = s;
    __syncthreads();
    if (t == 0) {
        float tot = 0.f;
        #pragma unroll
        for (int i = 0; i < 8; ++i) tot += red[i];
        ee[k] = tot;
        if (blockIdx.x == 0) out[LOSS_POS] = 0.0f;   // re-zeroed every launch
    }
}

// ---------------------------------------------------------------------------
// Kernel 2: bf16x3 split-precision distance GEMM + running argmin.
// fp32 x is stored as hi=bf16(x), lo=bf16(x-hi); dot = hi.hi + hi.lo + lo.hi
// with fp32 WMMA accumulation (error ~2^-32, argmin-safe). Per 16x16 tile per
// k-tile: 8 chunks x 3 = 24 v_wmma_f32_16x16x32_bf16 (vs 64 fp32 WMMAs).
// Grid: N/64 blocks of 128 threads (4 waves); each wave owns 16 z-rows.
// ---------------------------------------------------------------------------
__global__ void __launch_bounds__(128)
vq_argmin_kernel(const float* __restrict__ z,
                 const float* __restrict__ emb,
                 const float* __restrict__ ee,
                 int*   __restrict__ ws_idx,
                 float* __restrict__ out) {
    __shared__ __bf16 Ahi[64 * LSB], Alo[64 * LSB];      // 64 z-rows split
    __shared__ __bf16 Bhi[2][16 * LSB], Blo[2][16 * LSB]; // double-buffered B

    const int tid  = threadIdx.x;
    const int lane = tid & 31;
    const int wave = tid >> 5;          // 0..3
    const int half = lane >> 4;         // which 16-lane half
    const int l15  = lane & 15;

    const int nBase = blockIdx.x * 64;

    // ---- Stage + split A tile (z rows nBase..nBase+63), once ----
    for (int i = tid; i < 64 * D; i += 128) {
        const int r  = i >> 8;
        const int d  = i & 255;
        const int n0 = nBase + r;
        const int b  = n0 >> 10;
        const int hw = n0 & 1023;
        const float v = z[b * (D * HW) + d * HW + hw];
        const __bf16 h = (__bf16)v;
        Ahi[r * LSB + d] = h;
        Alo[r * LSB + d] = (__bf16)(v - (float)h);
    }

    // ---- Prologue: stage + split B tile for kt=0 into buffer 0 ----
    for (int i = tid; i < 16 * D; i += 128) {
        const int r = i >> 8;
        const int d = i & 255;
        const float v = emb[r * D + d];
        const __bf16 h = (__bf16)v;
        Bhi[0][r * LSB + d] = h;
        Blo[0][r * LSB + d] = (__bf16)(v - (float)h);
    }
    __syncthreads();

    float best[8];
    int   bidx[8];
    #pragma unroll
    for (int j = 0; j < 8; ++j) { best[j] = 3.4e38f; bidx[j] = 0; }

    const int aRow = wave * 16 + l15;   // A row this lane feeds
    int cur = 0;

    for (int kt = 0; kt < K; kt += 16) {
        // ---- Prefetch next B tile into registers (overlaps WMMA stream) ----
        const bool hasNext = (kt + 16) < K;
        float4 pf[8];
        if (hasNext) {
            #pragma unroll
            for (int j = 0; j < 8; ++j) {
                const int i  = tid + 128 * j;        // 0..1023 float4 slots
                const int r  = i >> 6;
                const int c4 = (i & 63) * 4;
                pf[j] = *(const float4*)&emb[(kt + 16 + r) * D + c4];
            }
        }

        // ---- 16x16 dot tile over D=256: 8 chunks x 3 bf16 WMMAs ----
        v8f acc = {};
        #pragma unroll
        for (int s = 0; s < D / 32; ++s) {
            // A (16x32 bf16, per-lane: two 8-elem chunks per §7.12.2)
            const int aoff = 32 * s + 8 * half;
            const __bf16* arow = &Ahi[aRow * LSB + aoff];
            v16bf ahi = cat8(*(const v8bf*)arow, *(const v8bf*)(arow + 16));
            const __bf16* arol = &Alo[aRow * LSB + aoff];
            v16bf alo = cat8(*(const v8bf*)arol, *(const v8bf*)(arol + 16));
            // B (32x16 bf16, per-lane: two 8-elem chunks, same k order as A)
            const int boff = 32 * s + 8 * half;
            const __bf16* brow = &Bhi[cur][l15 * LSB + boff];
            v16bf bhi = cat8(*(const v8bf*)brow, *(const v8bf*)(brow + 16));
            const __bf16* brol = &Blo[cur][l15 * LSB + boff];
            v16bf blo = cat8(*(const v8bf*)brol, *(const v8bf*)(brol + 16));

            acc = __builtin_amdgcn_wmma_f32_16x16x32_bf16(
                false, ahi, false, bhi, (short)0, acc, false, false);
            acc = __builtin_amdgcn_wmma_f32_16x16x32_bf16(
                false, ahi, false, blo, (short)0, acc, false, false);
            acc = __builtin_amdgcn_wmma_f32_16x16x32_bf16(
                false, alo, false, bhi, (short)0, acc, false, false);
        }

        // ---- Score and update argmin (||z||^2 dropped: row-constant) ----
        const int   kIdx = kt + l15;
        const float eeK  = ee[kIdx];
        #pragma unroll
        for (int j = 0; j < 8; ++j) {
            const float s = fmaf(-2.0f, acc[j], eeK);
            if (s < best[j]) { best[j] = s; bidx[j] = kIdx; }
        }

        // ---- Split + store prefetched tile into the other buffer ----
        if (hasNext) {
            const int nxt = cur ^ 1;
            #pragma unroll
            for (int j = 0; j < 8; ++j) {
                const int i = tid + 128 * j;
                const int r = i >> 6;
                const int c = (i & 63) * 4;
                const float vv[4] = {pf[j].x, pf[j].y, pf[j].z, pf[j].w};
                #pragma unroll
                for (int q = 0; q < 4; ++q) {
                    const __bf16 h = (__bf16)vv[q];
                    Bhi[nxt][r * LSB + c + q] = h;
                    Blo[nxt][r * LSB + c + q] = (__bf16)(vv[q] - (float)h);
                }
            }
        }
        __syncthreads();   // next iter reads buf nxt; also WAR-protects cur
        cur ^= 1;
    }

    // ---- Cross-lane argmin: xor masks 1..8 stay inside each 16-lane half,
    // reducing rows (wave*16+j) and (wave*16+j+8) simultaneously.
    #pragma unroll
    for (int j = 0; j < 8; ++j) {
        float v  = best[j];
        int   bi = bidx[j];
        #pragma unroll
        for (int m = 8; m >= 1; m >>= 1) {
            const float ov = __shfl_xor(v, m, 32);
            const int   oi = __shfl_xor(bi, m, 32);
            if (ov < v || (ov == v && oi < bi)) { v = ov; bi = oi; }
        }
        if (l15 == 0) {
            const int row = nBase + wave * 16 + (half ? (j + 8) : j);
            ws_idx[row]        = bi;
            out[IDX_POS + row] = (float)bi;   // indices as float in d_out
        }
    }
}

// ---------------------------------------------------------------------------
// Kernel 3: gather emb[idx] -> NCHW output + 1.25 * MSE loss accumulation.
// ---------------------------------------------------------------------------
__global__ void vq_gather_loss_kernel(const float* __restrict__ z,
                                      const float* __restrict__ emb,
                                      const int* __restrict__ ws_idx,
                                      float* __restrict__ out) {
    const int n  = blockIdx.x;
    const int d  = threadIdx.x;
    const int b  = n >> 10;
    const int hw = n & 1023;
    const int idx = ws_idx[n];

    const float e    = emb[idx * D + d];
    const int   zoff = b * (D * HW) + d * HW + hw;
    const float diff = e - z[zoff];
    out[zoff] = e;                       // straight-through value == quantized

    float s = diff * diff;
    #pragma unroll
    for (int m = 16; m >= 1; m >>= 1) s += __shfl_xor(s, m, 32);
    __shared__ float red[8];
    if ((d & 31) == 0) red[d >> 5] = s;
    __syncthreads();
    if (d == 0) {
        float tot = 0.f;
        #pragma unroll
        for (int i = 0; i < 8; ++i) tot += red[i];
        atomicAdd(&out[LOSS_POS], tot * (1.25f / (float)OUT_ELEMS));
    }
}

// ---------------------------------------------------------------------------
extern "C" void kernel_launch(void* const* d_in, const int* in_sizes, int n_in,
                              void* d_out, int out_size, void* d_ws, size_t ws_size,
                              hipStream_t stream) {
    const float* z   = (const float*)d_in[0];   // [B, D, H, W]
    const float* emb = (const float*)d_in[1];   // [K, D]
    float* out = (float*)d_out;

    float* ws_ee  = (float*)d_ws;               // K floats
    int*   ws_idx = (int*)d_ws + K;             // N ints

    vq_ee_kernel<<<K, 256, 0, stream>>>(emb, ws_ee, out);
    vq_argmin_kernel<<<N / 64, 128, 0, stream>>>(z, emb, ws_ee, ws_idx, out);
    vq_gather_loss_kernel<<<N, 256, 0, stream>>>(z, emb, ws_idx, out);
}
